// Network_13262859010083
// MI455X (gfx1250) — matrix operand order repaired and checked by reading, not compiled, not run on hardware
//
#include <hip/hip_runtime.h>
#include <stdint.h>
#include <stddef.h>

// ---------------------------------------------------------------------------
// Problem constants (from reference)
// ---------------------------------------------------------------------------
#define S_LEN 4096
#define EW    300
#define EP    100
#define DHID  400     // LSTM hidden per direction (= EW+EP)
#define K416  416     // DHID padded to multiple of 32 for WMMA K-loop
#define G4    1600    // 4*DHID gate width
#define D2    800     // 2*DHID
#define HID   512
#define FDIM  128

typedef __attribute__((ext_vector_type(16))) _Float16 v16h;
typedef __attribute__((ext_vector_type(8)))  float    v8f;
typedef __attribute__((ext_vector_type(4)))  uint32_t v4u;

static __device__ __forceinline__ float sigf(float x) {
  return 1.0f / (1.0f + __expf(-x));
}

// ---------------------------------------------------------------------------
// WMMA GEMM:  C[M,N] = A[M,K] @ B[N,K]^T  (+ col bias(es), + row bias)
// A, B f16 row-major; f32 accumulate; write f32 (Cf) and/or f16 (Ch).
// Each wave computes a 32x32 output block (2x2 WMMA tiles) so every A/B
// fragment feeds two v_wmma ops. Requirements (guaranteed by host code):
//   K % 32 == 0, M % 32 == 0, N % 32 == 0, rows 16-byte aligned.
// Fragment layouts per CDNA5 ISA 7.12.2 (wave32, 16-bit A 16x32 / B 32x16):
//   A: lane holds row (lane&15); dword i<4 -> K = akb+2i, i>=4 -> 16+akb+2(i-4)
//      => two contiguous 16B runs  -> 2x global_load_b128
//   B: lane holds col (lane&15); dword i<4 -> K = bkb+2i, i>=4 -> 16+bkb+2(i-4)
//      => two contiguous 16B runs  -> 2x global_load_b128
// ---------------------------------------------------------------------------
#define GEMM_WAVES 8

union Frag { v16h v; v4u q[2]; };

static __device__ __forceinline__ Frag load_a(const _Float16* p) {
  Frag r;
  r.q[0] = *(const v4u*)(p);        // K pairs akb .. akb+7
  r.q[1] = *(const v4u*)(p + 16);   // K pairs 16+akb .. 16+akb+7
  return r;
}
static __device__ __forceinline__ Frag load_b(const _Float16* p) {
  Frag r;
  r.q[0] = *(const v4u*)(p);        // K bkb .. bkb+7
  r.q[1] = *(const v4u*)(p + 16);   // K 16+bkb .. 16+bkb+7
  return r;
}

__global__ __launch_bounds__(32 * GEMM_WAVES)
void gemm_awt_f16(const _Float16* __restrict__ A, const _Float16* __restrict__ B,
                  float* __restrict__ Cf, _Float16* __restrict__ Ch,
                  const float* __restrict__ bias0, const float* __restrict__ bias1,
                  const float* __restrict__ rowbias,
                  int N, int K, int lda, int ldb, int ldc)
{
  const int lane  = threadIdx.x & 31;
  const int wave  = threadIdx.x >> 5;
  const int n0 = (blockIdx.x * GEMM_WAVES + wave) * 32;
  const int m0 = blockIdx.y * 32;
  if (n0 >= N) return;              // wave-uniform: EXEC all-ones for WMMA

  const int arow = lane & 15;
  const int akb  = (lane & 16) ? 8 : 0;
  const int bcol = lane & 15;
  const int bkb  = (lane & 16) ? 8 : 0;

  const _Float16* ap0 = A + (size_t)(m0 + arow)      * lda + akb;
  const _Float16* ap1 = A + (size_t)(m0 + 16 + arow) * lda + akb;
  const _Float16* bp0 = B + (size_t)(n0 + bcol)      * ldb + bkb;
  const _Float16* bp1 = B + (size_t)(n0 + 16 + bcol) * ldb + bkb;

  v8f acc00 = {}, acc01 = {}, acc10 = {}, acc11 = {};
  for (int k0 = 0; k0 < K; k0 += 32) {
    const Frag a0 = load_a(ap0 + k0);
    const Frag a1 = load_a(ap1 + k0);
    const Frag b0 = load_b(bp0 + k0);
    const Frag b1 = load_b(bp1 + k0);
    acc00 = __builtin_amdgcn_wmma_f32_16x16x32_f16(false, a0.v, false, b0.v,
                                                   (short)0, acc00, false, false);
    acc01 = __builtin_amdgcn_wmma_f32_16x16x32_f16(false, a0.v, false, b1.v,
                                                   (short)0, acc01, false, false);
    acc10 = __builtin_amdgcn_wmma_f32_16x16x32_f16(false, a1.v, false, b0.v,
                                                   (short)0, acc10, false, false);
    acc11 = __builtin_amdgcn_wmma_f32_16x16x32_f16(false, a1.v, false, b1.v,
                                                   (short)0, acc11, false, false);
  }

  // C layout: lane holds col (lane&15); VGPR r -> row r (+8 for hi half)
  const int cn = lane & 15;
  const int mo = (lane & 16) ? 8 : 0;
  float badd0 = 0.0f, badd1 = 0.0f;
  if (bias0) { badd0 += bias0[n0 + cn]; badd1 += bias0[n0 + 16 + cn]; }
  if (bias1) { badd0 += bias1[n0 + cn]; badd1 += bias1[n0 + 16 + cn]; }

  const v8f accs[2][2] = { { acc00, acc01 }, { acc10, acc11 } };
#pragma unroll
  for (int ti = 0; ti < 2; ++ti) {
#pragma unroll
    for (int r = 0; r < 8; ++r) {
      const int m = m0 + 16 * ti + mo + r;
      const float rbv = rowbias ? rowbias[m] : 0.0f;
      float v0 = accs[ti][0][r] + badd0 + rbv;
      float v1 = accs[ti][1][r] + badd1 + rbv;
      if (Cf) {
        Cf[(size_t)m * ldc + n0 + cn]      = v0;
        Cf[(size_t)m * ldc + n0 + 16 + cn] = v1;
      }
      if (Ch) {
        Ch[(size_t)m * ldc + n0 + cn]      = (_Float16)v0;
        Ch[(size_t)m * ldc + n0 + 16 + cn] = (_Float16)v1;
      }
    }
  }
}

// ---------------------------------------------------------------------------
// Persistent sequential LSTM scan: grid.x = 2 (dir 0 = fwd, 1 = bwd).
// G = precomputed x@Wih^T + bih + bhh  [S, 1600] (fp32, from WMMA GEMM).
// WT = Whh^T [400, 1600] fp32 (transposed for coalesced lane access).
// h, c, gate pre-activations live in LDS; two barriers per timestep.
// ---------------------------------------------------------------------------
__global__ __launch_bounds__(1024)
void lstm_seq(const float* __restrict__ Gf, const float* __restrict__ Gb,
              const float* __restrict__ WTf, const float* __restrict__ WTb,
              const float* __restrict__ h0, const float* __restrict__ c0,
              float* __restrict__ hf, float* __restrict__ hb)
{
  __shared__ float h_s[DHID];
  __shared__ float c_s[DHID];
  __shared__ float g_s[G4];

  const int dir = blockIdx.x;
  const float* G  = dir ? Gb  : Gf;
  const float* WT = dir ? WTb : WTf;
  float* hout     = dir ? hb  : hf;
  const int tid = threadIdx.x;

  if (tid < DHID) {
    h_s[tid] = h0[dir * DHID + tid];   // h0 flat [2,1,400]
    c_s[tid] = c0[dir * DHID + tid];
  }
  __syncthreads();

  for (int step = 0; step < S_LEN; ++step) {
    const int t = dir ? (S_LEN - 1 - step) : step;
    const float* grow = G + (size_t)t * G4;

    // gate pre-activations: g[j] = G[t,j] + sum_k Whh^T[k,j] * h[k]
    for (int j = tid; j < G4; j += 1024) {
      float acc = grow[j];
      const float* wp = WT + j;
#pragma unroll 4
      for (int k = 0; k < DHID; ++k) acc += wp[(size_t)k * G4] * h_s[k];
      g_s[j] = acc;
    }
    __syncthreads();

    // elementwise cell update, torch gate order (i, f, g, o)
    if (tid < DHID) {
      const float ig = sigf(g_s[tid]);
      const float fg = sigf(g_s[DHID + tid]);
      const float gg = tanhf(g_s[2 * DHID + tid]);
      const float og = sigf(g_s[3 * DHID + tid]);
      const float c2 = fg * c_s[tid] + ig * gg;
      const float h2 = og * tanhf(c2);
      c_s[tid] = c2;
      h_s[tid] = h2;
      hout[(size_t)t * DHID + tid] = h2;
    }
    __syncthreads();
  }
}

// ---------------------------------------------------------------------------
// Small prep / elementwise kernels
// ---------------------------------------------------------------------------
__global__ void f32_to_f16_k(const float* __restrict__ in, _Float16* __restrict__ out, int n)
{
  const int i = blockIdx.x * blockDim.x + threadIdx.x;
  if (i < n) out[i] = (_Float16)in[i];
}

// convert [rows, kin] f32 -> [rows, kout] f16 with zero K-padding
__global__ void f32_to_f16_padk_k(const float* __restrict__ in, _Float16* __restrict__ out,
                                  int rows, int kin, int kout)
{
  const int idx = blockIdx.x * blockDim.x + threadIdx.x;
  if (idx >= rows * kout) return;
  const int r = idx / kout;
  const int k = idx - r * kout;
  out[idx] = (k < kin) ? (_Float16)in[(size_t)r * kin + k] : (_Float16)0.0f;
}

__global__ void transpose_whh_k(const float* __restrict__ in, float* __restrict__ out)
{
  // in: [1600, 400] -> out: [400, 1600]
  const int idx = blockIdx.x * blockDim.x + threadIdx.x;
  if (idx >= G4 * DHID) return;
  const int j = idx / DHID;
  const int k = idx - j * DHID;
  out[(size_t)k * G4 + j] = in[idx];
}

__global__ void u1_transpose_f16_k(const float* __restrict__ U1, _Float16* __restrict__ U1t)
{
  // U1t[f][k] = U1[k][f]  (so T = Hh @ U1  ==  Hh @ U1t^T, one GEMM form)
  const int idx = blockIdx.x * blockDim.x + threadIdx.x;
  if (idx >= FDIM * FDIM) return;
  const int f = idx / FDIM;
  const int k = idx - f * FDIM;
  U1t[idx] = (_Float16)U1[(size_t)k * FDIM + f];
}

__global__ void embed_f16_k(const int* __restrict__ wi, const int* __restrict__ pi,
                            const float* __restrict__ wt, const float* __restrict__ pt,
                            _Float16* __restrict__ X)   // X: [S, K416], zero-padded
{
  const int idx = blockIdx.x * blockDim.x + threadIdx.x;
  if (idx >= S_LEN * K416) return;
  const int t = idx / K416;
  const int d = idx - t * K416;
  float v = 0.0f;
  if (d < EW)        v = wt[(size_t)wi[t] * EW + d];
  else if (d < DHID) v = pt[(size_t)pi[t] * EP + (d - EW)];
  X[idx] = (_Float16)v;
}

__global__ void sig_concat_k(const float* __restrict__ hf, const float* __restrict__ hb,
                             _Float16* __restrict__ S8)
{
  const int idx = blockIdx.x * blockDim.x + threadIdx.x;
  if (idx >= S_LEN * D2) return;
  const int t = idx / D2;
  const int d = idx - t * D2;
  const float v = (d < DHID) ? hf[(size_t)t * DHID + d]
                             : hb[(size_t)t * DHID + (d - DHID)];
  S8[idx] = (_Float16)sigf(v);
}

__global__ void rowbias_k(const _Float16* __restrict__ Hh, const float* __restrict__ u2,
                          float* __restrict__ rb)
{
  const int m = blockIdx.x * blockDim.x + threadIdx.x;
  if (m >= S_LEN) return;
  float a = 0.0f;
  const _Float16* hp = Hh + (size_t)m * FDIM;
#pragma unroll 4
  for (int f = 0; f < FDIM; ++f) a += (float)hp[f] * u2[f];
  rb[m] = a;
}

// ---------------------------------------------------------------------------
// Host orchestration
// ---------------------------------------------------------------------------
static inline dim3 gemm_grid(int M, int N) {
  const int npairs = N / 32;        // all N here are multiples of 32
  return dim3((npairs + GEMM_WAVES - 1) / GEMM_WAVES, M / 32, 1);
}

extern "C" void kernel_launch(void* const* d_in, const int* in_sizes, int n_in,
                              void* d_out, int out_size, void* d_ws, size_t ws_size,
                              hipStream_t stream)
{
  (void)in_sizes; (void)n_in; (void)out_size; (void)ws_size;

  const int*   wi    = (const int*)  d_in[0];
  const int*   pi    = (const int*)  d_in[1];
  const float* h0    = (const float*)d_in[2];
  const float* c0    = (const float*)d_in[3];
  const float* wt    = (const float*)d_in[4];
  const float* pt    = (const float*)d_in[5];
  const float* Wih_f = (const float*)d_in[6];
  const float* Whh_f = (const float*)d_in[7];
  const float* bih_f = (const float*)d_in[8];
  const float* bhh_f = (const float*)d_in[9];
  const float* Wih_b = (const float*)d_in[10];
  const float* Whh_b = (const float*)d_in[11];
  const float* bih_b = (const float*)d_in[12];
  const float* bhh_b = (const float*)d_in[13];
  const float* Wh1   = (const float*)d_in[14];
  const float* bh1   = (const float*)d_in[15];
  const float* Wh2   = (const float*)d_in[16];
  const float* bh2   = (const float*)d_in[17];
  const float* Wd1   = (const float*)d_in[18];
  const float* bd1   = (const float*)d_in[19];
  const float* Wd2   = (const float*)d_in[20];
  const float* bd2   = (const float*)d_in[21];
  const float* U1    = (const float*)d_in[22];
  const float* u2    = (const float*)d_in[23];
  float* out = (float*)d_out;

  // bump allocator on preallocated workspace (~96 MB total)
  char* ws = (char*)d_ws;
  size_t off = 0;
  auto alloc = [&](size_t bytes) -> void* {
    void* p = ws + off;
    off = (off + bytes + 255) & ~(size_t)255;
    return p;
  };

  _Float16* Xh     = (_Float16*)alloc((size_t)S_LEN * K416 * 2);   // K-padded
  _Float16* WihFh  = (_Float16*)alloc((size_t)G4 * K416 * 2);      // K-padded
  _Float16* WihBh  = (_Float16*)alloc((size_t)G4 * K416 * 2);      // K-padded
  float*    Gfws   = (float*)   alloc((size_t)S_LEN * G4 * 4);
  float*    Gbws   = (float*)   alloc((size_t)S_LEN * G4 * 4);
  float*    WTf    = (float*)   alloc((size_t)DHID * G4 * 4);
  float*    WTb    = (float*)   alloc((size_t)DHID * G4 * 4);
  float*    hfws   = (float*)   alloc((size_t)S_LEN * DHID * 4);
  float*    hbws   = (float*)   alloc((size_t)S_LEN * DHID * 4);
  _Float16* S8     = (_Float16*)alloc((size_t)S_LEN * D2 * 2);
  _Float16* Wh1h   = (_Float16*)alloc((size_t)HID * D2 * 2);
  _Float16* Wd1h   = (_Float16*)alloc((size_t)HID * D2 * 2);
  _Float16* Wh2h   = (_Float16*)alloc((size_t)FDIM * HID * 2);
  _Float16* Wd2h   = (_Float16*)alloc((size_t)FDIM * HID * 2);
  _Float16* H1h    = (_Float16*)alloc((size_t)S_LEN * HID * 2);
  _Float16* H1d    = (_Float16*)alloc((size_t)S_LEN * HID * 2);
  _Float16* Hh     = (_Float16*)alloc((size_t)S_LEN * FDIM * 2);
  _Float16* Hd     = (_Float16*)alloc((size_t)S_LEN * FDIM * 2);
  _Float16* U1t    = (_Float16*)alloc((size_t)FDIM * FDIM * 2);
  _Float16* Tm     = (_Float16*)alloc((size_t)S_LEN * FDIM * 2);
  float*    rb     = (float*)   alloc((size_t)S_LEN * 4);

  const int TB = 256;
  auto nb = [](int n, int tb) { return (n + tb - 1) / tb; };

  // --- prep: weight converts / transposes ---------------------------------
  f32_to_f16_padk_k<<<nb(G4 * K416, TB), TB, 0, stream>>>(Wih_f, WihFh, G4, DHID, K416);
  f32_to_f16_padk_k<<<nb(G4 * K416, TB), TB, 0, stream>>>(Wih_b, WihBh, G4, DHID, K416);
  f32_to_f16_k<<<nb(HID * D2,  TB), TB, 0, stream>>>(Wh1, Wh1h, HID * D2);
  f32_to_f16_k<<<nb(HID * D2,  TB), TB, 0, stream>>>(Wd1, Wd1h, HID * D2);
  f32_to_f16_k<<<nb(FDIM * HID, TB), TB, 0, stream>>>(Wh2, Wh2h, FDIM * HID);
  f32_to_f16_k<<<nb(FDIM * HID, TB), TB, 0, stream>>>(Wd2, Wd2h, FDIM * HID);
  transpose_whh_k<<<nb(G4 * DHID, TB), TB, 0, stream>>>(Whh_f, WTf);
  transpose_whh_k<<<nb(G4 * DHID, TB), TB, 0, stream>>>(Whh_b, WTb);
  u1_transpose_f16_k<<<nb(FDIM * FDIM, TB), TB, 0, stream>>>(U1, U1t);

  // --- embeddings (zero K-padded) ------------------------------------------
  embed_f16_k<<<nb(S_LEN * K416, TB), TB, 0, stream>>>(wi, pi, wt, pt, Xh);

  // --- input projections (WMMA): G = X @ Wih^T + bih + bhh ----------------
  gemm_awt_f16<<<gemm_grid(S_LEN, G4), 32 * GEMM_WAVES, 0, stream>>>(
      Xh, WihFh, Gfws, nullptr, bih_f, bhh_f, nullptr,
      G4, K416, K416, K416, G4);
  gemm_awt_f16<<<gemm_grid(S_LEN, G4), 32 * GEMM_WAVES, 0, stream>>>(
      Xh, WihBh, Gbws, nullptr, bih_b, bhh_b, nullptr,
      G4, K416, K416, K416, G4);

  // --- sequential bidirectional LSTM scan ---------------------------------
  lstm_seq<<<2, 1024, 0, stream>>>(Gfws, Gbws, WTf, WTb, h0, c0, hfws, hbws);

  // --- sigmoid(concat) -----------------------------------------------------
  sig_concat_k<<<nb(S_LEN * D2, TB), TB, 0, stream>>>(hfws, hbws, S8);

  // --- MLP heads (WMMA) ----------------------------------------------------
  gemm_awt_f16<<<gemm_grid(S_LEN, HID), 32 * GEMM_WAVES, 0, stream>>>(
      S8, Wh1h, nullptr, H1h, bh1, nullptr, nullptr,
      HID, D2, D2, D2, HID);
  gemm_awt_f16<<<gemm_grid(S_LEN, FDIM), 32 * GEMM_WAVES, 0, stream>>>(
      H1h, Wh2h, nullptr, Hh, bh2, nullptr, nullptr,
      FDIM, HID, HID, HID, FDIM);
  gemm_awt_f16<<<gemm_grid(S_LEN, HID), 32 * GEMM_WAVES, 0, stream>>>(
      S8, Wd1h, nullptr, H1d, bd1, nullptr, nullptr,
      HID, D2, D2, D2, HID);
  gemm_awt_f16<<<gemm_grid(S_LEN, FDIM), 32 * GEMM_WAVES, 0, stream>>>(
      H1d, Wd2h, nullptr, Hd, bd2, nullptr, nullptr,
      FDIM, HID, HID, HID, FDIM);

  // --- biaffine ------------------------------------------------------------
  // T = h_head @ U1  (== Hh @ U1t^T), WMMA
  gemm_awt_f16<<<gemm_grid(S_LEN, FDIM), 32 * GEMM_WAVES, 0, stream>>>(
      Hh, U1t, nullptr, Tm, nullptr, nullptr, nullptr,
      FDIM, FDIM, FDIM, FDIM, FDIM);
  // rowbias[m] = h_head[m] . u2
  rowbias_k<<<nb(S_LEN, TB), TB, 0, stream>>>(Hh, u2, rb);
  // adj = T @ h_dep^T + rowbias  -> d_out (fp32)
  gemm_awt_f16<<<gemm_grid(S_LEN, S_LEN), 32 * GEMM_WAVES, 0, stream>>>(
      Tm, Hd, out, nullptr, nullptr, nullptr, rb,
      S_LEN, FDIM, FDIM, FDIM, S_LEN);
}
